// EconomicRelationshipDiscoverer_40750649705020
// MI455X (gfx1250) — matrix operand order repaired and checked by reading, not compiled, not run on hardware
//
#include <hip/hip_runtime.h>
#include <hip/hip_bf16.h>
#include <math.h>

typedef __attribute__((ext_vector_type(16))) _Float16 v16h;
typedef __attribute__((ext_vector_type(8)))  _Float16 v8h;
typedef __attribute__((ext_vector_type(8)))  float    v8f;

#define BB 4
#define SS 256
#define NN 500
#define FIN 32
#define DD 64
#define MROWS (BB*SS*NN)      // 512000
#define NDFLAT (NN*DD)        // 32000
#define NPAD 512
#define BSN (BB*SS)           // 1024

__device__ __forceinline__ float gelu_erf(float x) {
  return 0.5f * x * (1.0f + erff(x * 0.7071067811865476f));
}
__device__ __forceinline__ float sigmf(float x) {
  return 1.0f / (1.0f + expf(-x));
}
__device__ __forceinline__ v8f wmma16(v16h a, v16h b, v8f c) {
  return __builtin_amdgcn_wmma_f32_16x16x32_f16(false, a, false, b, (short)0, c,
                                                false, false);
}
// A fragment (16x32 f16): two 16B chunks at +0 and +8 halves.
__device__ __forceinline__ v16h load_afrag(const _Float16* base) {
  v8h lo = *(const v8h*)(base);
  v8h hi = *(const v8h*)(base + 8);
  union { v16h v; v8h h[2]; } u;
  u.h[0] = lo; u.h[1] = hi;
  return u.v;
}
// B fragment (32x16 f16) when the 16 K-halves for this lane are contiguous.
__device__ __forceinline__ v16h load_bfrag(const _Float16* base) {
  v8h lo = *(const v8h*)(base);
  v8h hi = *(const v8h*)(base + 8);
  union { v16h v; v8h h[2]; } u;
  u.h[0] = lo; u.h[1] = hi;
  return u.v;
}

// ---------------- small utility kernels ----------------
__global__ void k_zero_h(_Float16* d, int n) {
  int i = blockIdx.x * blockDim.x + threadIdx.x;
  if (i < n) d[i] = (_Float16)0.f;
}
__global__ void k_cvt_h(const float* __restrict__ s, _Float16* __restrict__ d, int n) {
  int i = blockIdx.x * blockDim.x + threadIdx.x;
  if (i < n) d[i] = (_Float16)s[i];
}

// adjT[m*512 + n] = softmax_row_n(from@to^T / 0.5)[m], zero-padded to 512x512
__global__ void k_adj(const float* __restrict__ fe, const float* __restrict__ te,
                      _Float16* __restrict__ adjT) {
  int row = blockIdx.x;        // source node n
  int tid = threadIdx.x;       // 64 threads
  __shared__ float frow[32];
  __shared__ float red[64];
  if (tid < 32) frow[tid] = fe[row * 32 + tid];
  __syncthreads();
  float lg[8];
  float mx = -1e30f;
#pragma unroll
  for (int i = 0; i < 8; ++i) {
    int c = tid + i * 64;
    float d = -1e30f;
    if (c < NN) {
      d = 0.f;
      for (int j = 0; j < 32; ++j) d += frow[j] * te[c * 32 + j];
      d *= 2.0f;  // / TEMP(0.5)
    }
    lg[i] = d;
    mx = fmaxf(mx, d);
  }
  red[tid] = mx; __syncthreads();
  for (int off = 32; off > 0; off >>= 1) {
    if (tid < off) red[tid] = fmaxf(red[tid], red[tid + off]);
    __syncthreads();
  }
  mx = red[0]; __syncthreads();
  float s = 0.f;
#pragma unroll
  for (int i = 0; i < 8; ++i) {
    int c = tid + i * 64;
    if (c < NN) { float e = expf(lg[i] - mx); lg[i] = e; s += e; }
  }
  red[tid] = s; __syncthreads();
  for (int off = 32; off > 0; off >>= 1) {
    if (tid < off) red[tid] += red[tid + off];
    __syncthreads();
  }
  float inv = 1.f / red[0];
#pragma unroll
  for (int i = 0; i < 8; ++i) {
    int c = tid + i * 64;
    if (c < NN) adjT[(size_t)c * NPAD + row] = (_Float16)(lg[i] * inv);
  }
}

// ------- WMMA GEMM, Ncols=64: C = act(A[MxK] @ W^T + bias), W f16 [64 x K] -------
template <int K, bool LN>
__global__ void k_gemm_nc64(const _Float16* __restrict__ A,
                            const _Float16* __restrict__ Wh,   // [64 x K] row-major
                            const float* __restrict__ bias,
                            const float* __restrict__ gvec,
                            const float* __restrict__ bevec,
                            float* __restrict__ outf,
                            _Float16* __restrict__ outh_t,
                            _Float16* __restrict__ outh_r, int M) {
  int lane = threadIdx.x & 31;
  int wave = threadIdx.x >> 5;
  long m0 = ((long)blockIdx.x * 8 + wave) * 16;
  if (m0 >= M) return;
  int lh = (lane >= 16) ? 1 : 0;
  int l15 = lane & 15;
  v8f acc[4];
#pragma unroll
  for (int t = 0; t < 4; ++t) acc[t] = (v8f)(0.f);
  const long arow = (m0 + l15) * (long)K;
#pragma unroll
  for (int ks = 0; ks < (K >> 5); ++ks) {
    v16h af = load_afrag(A + arow + ks * 32 + lh * 16);
#pragma unroll
    for (int t = 0; t < 4; ++t) {
      int col = t * 16 + l15;
      v16h bf = load_bfrag(Wh + col * K + ks * 32 + lh * 16);
      acc[t] = wmma16(af, bf, acc[t]);
    }
  }
  float vals[4][8];
#pragma unroll
  for (int t = 0; t < 4; ++t) {
    float bi = bias[t * 16 + l15];
#pragma unroll
    for (int r = 0; r < 8; ++r) vals[t][r] = gelu_erf(acc[t][r] + bi);
  }
  if (LN) {
#pragma unroll
    for (int r = 0; r < 8; ++r) {
      float s = vals[0][r] + vals[1][r] + vals[2][r] + vals[3][r];
      s += __shfl_xor(s, 1, 32); s += __shfl_xor(s, 2, 32);
      s += __shfl_xor(s, 4, 32); s += __shfl_xor(s, 8, 32);
      float mu = s * (1.f / 64.f);
      float q = 0.f;
#pragma unroll
      for (int t = 0; t < 4; ++t) { float d = vals[t][r] - mu; q += d * d; }
      q += __shfl_xor(q, 1, 32); q += __shfl_xor(q, 2, 32);
      q += __shfl_xor(q, 4, 32); q += __shfl_xor(q, 8, 32);
      float rstd = rsqrtf(q * (1.f / 64.f) + 1e-5f);
#pragma unroll
      for (int t = 0; t < 4; ++t) vals[t][r] = (vals[t][r] - mu) * rstd;
    }
#pragma unroll
    for (int t = 0; t < 4; ++t) {
      float gg = gvec[t * 16 + l15], bb2 = bevec[t * 16 + l15];
#pragma unroll
      for (int r = 0; r < 8; ++r) vals[t][r] = vals[t][r] * gg + bb2;
    }
  }
#pragma unroll
  for (int r = 0; r < 8; ++r) {
    unsigned row = (unsigned)(m0 + r + lh * 8);
#pragma unroll
    for (int t = 0; t < 4; ++t) {
      int col = t * 16 + l15;
      if (outf) outf[(size_t)row * 64 + col] = vals[t][r];
      if (outh_r) outh_r[(size_t)row * 64 + col] = (_Float16)vals[t][r];
    }
    if (outh_t) {
      unsigned bs = row / NN;
      unsigned n = row - bs * NN;
#pragma unroll
      for (int t = 0; t < 4; ++t) {
        int col = t * 16 + l15;
        outh_t[(size_t)bs * 64 * NPAD + (size_t)col * NPAD + n] =
            (_Float16)vals[t][r];
      }
    }
  }
}

// ------- aggregation: per (b,s): y = adj^T @ X + x, X transposed-padded ----------
// double-buffered fragment pipeline
__global__ void k_agg(const _Float16* __restrict__ adjT,   // [512 x 512]
                      const _Float16* __restrict__ xt,     // [1024][64][512]
                      const float* __restrict__ x0,        // row-major
                      _Float16* __restrict__ yh) {         // row-major f16
  int lane = threadIdx.x & 31;
  int wave = threadIdx.x >> 5;
  int bs = blockIdx.y;
  int m0 = (blockIdx.x * 8 + wave) * 16;  // 0..496 (32 m-tiles)
  int lh = (lane >= 16) ? 1 : 0;
  int l15 = lane & 15;
  const _Float16* xtb = xt + (size_t)bs * 64 * NPAD;
  const float* x0b = x0 + (size_t)bs * NDFLAT;
  _Float16* yb = yh + (size_t)bs * NDFLAT;
  v8f acc[4];
#pragma unroll
  for (int t = 0; t < 4; ++t) acc[t] = (v8f)(0.f);
  const _Float16* abase = adjT + (size_t)(m0 + l15) * NPAD + lh * 16;
  const _Float16* bbase = xtb + (size_t)l15 * NPAD + lh * 16;

  v16h a_cur = load_afrag(abase);
  v16h b_cur[4];
#pragma unroll
  for (int t = 0; t < 4; ++t) b_cur[t] = load_bfrag(bbase + (size_t)t * 16 * NPAD);
  for (int ks = 0; ks < 16; ++ks) {
    int ksn = (ks + 1 < 16) ? ks + 1 : ks;
    v16h a_nxt = load_afrag(abase + ksn * 32);
    v16h b_nxt[4];
#pragma unroll
    for (int t = 0; t < 4; ++t)
      b_nxt[t] = load_bfrag(bbase + (size_t)t * 16 * NPAD + ksn * 32);
#pragma unroll
    for (int t = 0; t < 4; ++t) acc[t] = wmma16(a_cur, b_cur[t], acc[t]);
    a_cur = a_nxt;
#pragma unroll
    for (int t = 0; t < 4; ++t) b_cur[t] = b_nxt[t];
  }
#pragma unroll
  for (int r = 0; r < 8; ++r) {
    int row = m0 + r + lh * 8;
    if (row < NN) {
#pragma unroll
      for (int t = 0; t < 4; ++t) {
        int idx = row * 64 + t * 16 + l15;
        yb[idx] = (_Float16)(acc[t][r] + x0b[idx]);
      }
    }
  }
}

// ------- GRU input projection partials: xpart = x_flat[slice] @ Wih^T -------------
// one wave per block; blockIdx.y = 64-col group (of 192); blockIdx.z = K-part
__global__ void k_gru_in(const _Float16* __restrict__ xflat,  // [1024 x 32000]
                         const _Float16* __restrict__ Wh,     // [192 x 32000] f16
                         float* __restrict__ xpart, int Tc, int Ts, int Mvalid,
                         int Mpad, int stepsPerPart) {
  int lane = threadIdx.x & 31;
  int lh = (lane >= 16) ? 1 : 0;
  int l15 = lane & 15;
  int colg = blockIdx.y;
  int part = blockIdx.z;
  int m0 = blockIdx.x * 16;
  int r = m0 + l15;
  int arow = 0;
  if (r < Mvalid) { int b = r / Tc; int t = Ts + r % Tc; arow = b * SS + t; }
  const _Float16* abase = xflat + (size_t)arow * NDFLAT + lh * 16;
  const _Float16* bbase = Wh + (size_t)(colg * 64 + l15) * NDFLAT + lh * 16;
  float* outp = xpart + (size_t)part * Mpad * 192;
  const int ks0 = part * stepsPerPart;
  const int ksEnd = ks0 + stepsPerPart;
  v8f acc[4];
#pragma unroll
  for (int t = 0; t < 4; ++t) acc[t] = (v8f)(0.f);

  v16h a_cur = load_afrag(abase + (size_t)ks0 * 32);
  v16h b_cur[4];
#pragma unroll
  for (int t = 0; t < 4; ++t)
    b_cur[t] = load_bfrag(bbase + (size_t)t * 16 * NDFLAT + (size_t)ks0 * 32);
  for (int ks = ks0; ks < ksEnd; ++ks) {
    int ksn = (ks + 1 < ksEnd) ? ks + 1 : ks;
    __builtin_prefetch(abase + (size_t)ksn * 32 + 512, 0, 1);
    v16h a_nxt = load_afrag(abase + (size_t)ksn * 32);
    v16h b_nxt[4];
#pragma unroll
    for (int t = 0; t < 4; ++t)
      b_nxt[t] = load_bfrag(bbase + (size_t)t * 16 * NDFLAT + (size_t)ksn * 32);
#pragma unroll
    for (int t = 0; t < 4; ++t) acc[t] = wmma16(a_cur, b_cur[t], acc[t]);
    a_cur = a_nxt;
#pragma unroll
    for (int t = 0; t < 4; ++t) b_cur[t] = b_nxt[t];
  }
#pragma unroll
  for (int rr = 0; rr < 8; ++rr) {
    int row = m0 + rr + lh * 8;
    if (row < Mvalid) {
#pragma unroll
      for (int t = 0; t < 4; ++t) {
        int col = colg * 64 + t * 16 + l15;
        outp[(size_t)row * 192 + col] = acc[t][rr];
      }
    }
  }
}

// deterministic fixed-order reduction of K-parts + bias
__global__ void k_xp_red(const float* __restrict__ xpart,
                         const float* __restrict__ bih, float* __restrict__ xp,
                         int Mvalid, int Mpad, int P) {
  int idx = blockIdx.x * blockDim.x + threadIdx.x;
  if (idx >= Mvalid * 192) return;
  int col = idx % 192;
  float s = bih[col];
  for (int pp = 0; pp < P; ++pp) s += xpart[(size_t)pp * Mpad * 192 + idx];
  xp[idx] = s;
}

// ------- GRU recurrence (sequential over T), 1 block ------------------------------
__global__ void k_gru(const float* __restrict__ xp, const float* __restrict__ Whh,
                      const float* __restrict__ bhh, float* __restrict__ hout, int T) {
  __shared__ float whh[192 * 64];
  __shared__ float h[256];
  __shared__ float gh[768];
  int tid = threadIdx.x;
  for (int i = tid; i < 192 * 64; i += 256) whh[i] = Whh[i];
  h[tid] = 0.f;
  __syncthreads();
  int b = tid >> 6, d = tid & 63;
  for (int t = 0; t < T; ++t) {
    for (int idx = tid; idx < 768; idx += 256) {
      int bb2 = idx / 192, g = idx % 192;
      float s = bhh[g];
      const float* wr = &whh[g * 64];
      const float* hb = &h[bb2 * 64];
      for (int dd = 0; dd < 64; ++dd) s += wr[dd] * hb[dd];
      gh[idx] = s;
    }
    __syncthreads();
    const float* xrow = xp + (size_t)(b * T + t) * 192;
    float rg = sigmf(xrow[d] + gh[b * 192 + d]);
    float z = sigmf(xrow[64 + d] + gh[b * 192 + 64 + d]);
    float n = tanhf(xrow[128 + d] + rg * gh[b * 192 + 128 + d]);
    float hnew = (1.f - z) * n + z * h[tid];
    __syncthreads();
    h[tid] = hnew;
    __syncthreads();
  }
  hout[tid] = h[tid];
}

// ------- decision head -------------------------------------------------------------
__global__ void k_head(const float* __restrict__ hs, const float* __restrict__ hm,
                       const float* __restrict__ hl, const float* __restrict__ W1,
                       const float* __restrict__ b1, const float* __restrict__ W2,
                       const float* __restrict__ b2, float* __restrict__ out) {
  __shared__ float hc[768];
  __shared__ float h1[256];
  int tid = threadIdx.x;
  for (int i = tid; i < 768; i += 256) {
    int b = i / 192, j = i % 192;
    float v = (j < 64) ? hs[b * 64 + j]
              : (j < 128) ? hm[b * 64 + (j - 64)] : hl[b * 64 + (j - 128)];
    hc[i] = v;
  }
  __syncthreads();
  {
    int b = tid >> 6, d = tid & 63;
    float s = b1[d];
    for (int j = 0; j < 192; ++j) s += W1[d * 192 + j] * hc[b * 192 + j];
    h1[tid] = gelu_erf(s);
  }
  __syncthreads();
  for (int o = tid; o < BB * NN; o += 256) {
    int b = o / NN, nn = o % NN;
    float s = b2[nn];
    for (int d = 0; d < 64; ++d) s += W2[nn * 64 + d] * h1[b * 64 + d];
    out[o] = s;
  }
}

extern "C" void kernel_launch(void* const* d_in, const int* in_sizes, int n_in,
                              void* d_out, int out_size, void* d_ws, size_t ws_size,
                              hipStream_t stream) {
  (void)in_sizes; (void)n_in; (void)out_size; (void)ws_size;
  const float* x     = (const float*)d_in[0];
  const float* ip_W  = (const float*)d_in[1];
  const float* ip_b  = (const float*)d_in[2];
  const float* fe    = (const float*)d_in[3];
  const float* te    = (const float*)d_in[4];
  const float* gl_W  = (const float*)d_in[5];
  const float* gl_b  = (const float*)d_in[6];
  const float* gl_g  = (const float*)d_in[7];
  const float* gl_be = (const float*)d_in[8];
  const float* Wih_s = (const float*)d_in[9];
  const float* Whh_s = (const float*)d_in[10];
  const float* bih_s = (const float*)d_in[11];
  const float* bhh_s = (const float*)d_in[12];
  const float* Wih_m = (const float*)d_in[13];
  const float* Whh_m = (const float*)d_in[14];
  const float* bih_m = (const float*)d_in[15];
  const float* bhh_m = (const float*)d_in[16];
  const float* Wih_l = (const float*)d_in[17];
  const float* Whh_l = (const float*)d_in[18];
  const float* bih_l = (const float*)d_in[19];
  const float* bhh_l = (const float*)d_in[20];
  const float* dh_W1 = (const float*)d_in[21];
  const float* dh_b1 = (const float*)d_in[22];
  const float* dh_W2 = (const float*)d_in[23];
  const float* dh_b2 = (const float*)d_in[24];
  float* out = (float*)d_out;

  char* p = (char*)d_ws;
  auto alloc = [&](size_t n) -> char* {
    char* r = p; p += (n + 255) & ~(size_t)255; return r;
  };
  _Float16* adjT   = (_Float16*)alloc((size_t)NPAD * NPAD * 2);
  _Float16* xAh    = (_Float16*)alloc((size_t)MROWS * FIN * 2);
  _Float16* ipWh   = (_Float16*)alloc((size_t)64 * 32 * 2);
  _Float16* g0h    = (_Float16*)alloc((size_t)64 * 64 * 2);
  _Float16* g1h    = (_Float16*)alloc((size_t)64 * 64 * 2);
  _Float16* Wsh    = (_Float16*)alloc((size_t)192 * NDFLAT * 2);
  _Float16* Wmh    = (_Float16*)alloc((size_t)192 * NDFLAT * 2);
  _Float16* Wlh    = (_Float16*)alloc((size_t)192 * NDFLAT * 2);
  float*    x0     = (float*)alloc((size_t)MROWS * 64 * 4);
  _Float16* xt     = (_Float16*)alloc((size_t)BSN * 64 * NPAD * 2);  // transposed+pad
  _Float16* yh     = (_Float16*)alloc((size_t)MROWS * 64 * 2);       // row-major
  _Float16* xflat  = (_Float16*)alloc((size_t)MROWS * 64 * 2);       // row-major
  float*    xpartS = (float*)alloc((size_t)1 * 48 * 192 * 4);
  float*    xpartM = (float*)alloc((size_t)1 * 80 * 192 * 4);
  float*    xpartL = (float*)alloc((size_t)4 * 1024 * 192 * 4);
  float*    xps    = (float*)alloc((size_t)48 * 192 * 4);
  float*    xpm    = (float*)alloc((size_t)80 * 192 * 4);
  float*    xpl    = (float*)alloc((size_t)1024 * 192 * 4);
  float*    hsO    = (float*)alloc(256 * 4);
  float*    hmO    = (float*)alloc(256 * 4);
  float*    hlO    = (float*)alloc(256 * 4);

  // adjacency (zero pad first) + zero xt pads (pads never rewritten)
  k_zero_h<<<(NPAD * NPAD + 255) / 256, 256, 0, stream>>>(adjT, NPAD * NPAD);
  {
    int nxt = BSN * 64 * NPAD;
    k_zero_h<<<(nxt + 255) / 256, 256, 0, stream>>>(xt, nxt);
  }
  k_adj<<<NN, 64, 0, stream>>>(fe, te, adjT);
  // f16 conversions (no transposes needed: weight rows are already K-contiguous)
  k_cvt_h<<<(MROWS * FIN + 255) / 256, 256, 0, stream>>>(x, xAh, MROWS * FIN);
  k_cvt_h<<<(64 * 32 + 255) / 256, 256, 0, stream>>>(ip_W, ipWh, 64 * 32);
  k_cvt_h<<<(64 * 64 + 255) / 256, 256, 0, stream>>>(gl_W, g0h, 64 * 64);
  k_cvt_h<<<(64 * 64 + 255) / 256, 256, 0, stream>>>(gl_W + 64 * 64, g1h, 64 * 64);
  int wn = 192 * NDFLAT;
  k_cvt_h<<<(wn + 255) / 256, 256, 0, stream>>>(Wih_s, Wsh, wn);
  k_cvt_h<<<(wn + 255) / 256, 256, 0, stream>>>(Wih_m, Wmh, wn);
  k_cvt_h<<<(wn + 255) / 256, 256, 0, stream>>>(Wih_l, Wlh, wn);

  // input projection: x0 (f32 rm) + xt (f16 transposed) = gelu(x @ ipW^T + b)
  k_gemm_nc64<32, false><<<MROWS / 128, 256, 0, stream>>>(
      xAh, ipWh, ip_b, nullptr, nullptr, x0, xt, nullptr, MROWS);
  // graph layer 0
  k_agg<<<dim3(4, BSN), 256, 0, stream>>>(adjT, xt, x0, yh);
  k_gemm_nc64<64, true><<<MROWS / 128, 256, 0, stream>>>(
      yh, g0h, gl_b, gl_g, gl_be, x0, xt, nullptr, MROWS);
  // graph layer 1
  k_agg<<<dim3(4, BSN), 256, 0, stream>>>(adjT, xt, x0, yh);
  k_gemm_nc64<64, true><<<MROWS / 128, 256, 0, stream>>>(
      yh, g1h, gl_b + 64, gl_g + 64, gl_be + 64, nullptr, nullptr, xflat, MROWS);
  // GRU input projections (xflat == x_flat f16 [1024 x 32000])
  k_gru_in<<<dim3(3, 3, 1), 32, 0, stream>>>(xflat, Wsh, xpartS, 10, SS - 10, 40,
                                             48, NDFLAT / 32);
  k_gru_in<<<dim3(5, 3, 1), 32, 0, stream>>>(xflat, Wmh, xpartM, 20, SS - 20, 80,
                                             80, NDFLAT / 32);
  k_gru_in<<<dim3(64, 3, 4), 32, 0, stream>>>(xflat, Wlh, xpartL, 256, 0, 1024,
                                              1024, NDFLAT / 32 / 4);
  k_xp_red<<<(40 * 192 + 255) / 256, 256, 0, stream>>>(xpartS, bih_s, xps, 40, 48, 1);
  k_xp_red<<<(80 * 192 + 255) / 256, 256, 0, stream>>>(xpartM, bih_m, xpm, 80, 80, 1);
  k_xp_red<<<(1024 * 192 + 255) / 256, 256, 0, stream>>>(xpartL, bih_l, xpl, 1024,
                                                         1024, 4);
  // GRU recurrences
  k_gru<<<1, 256, 0, stream>>>(xps, Whh_s, bhh_s, hsO, 10);
  k_gru<<<1, 256, 0, stream>>>(xpm, Whh_m, bhh_m, hmO, 20);
  k_gru<<<1, 256, 0, stream>>>(xpl, Whh_l, bhh_l, hlO, 256);
  // decision head
  k_head<<<1, 256, 0, stream>>>(hsO, hmO, hlO, dh_W1, dh_b1, dh_W2, dh_b2, out);
}
